// MultiScaleRouter_12575664243336
// MI455X (gfx1250) — matrix operand order repaired and checked by reading, not compiled, not run on hardware
//
#include <hip/hip_runtime.h>
#include <hip/hip_bf16.h>
#include <math.h>

// ---------------- CDNA5 WMMA types ----------------
typedef __attribute__((ext_vector_type(16))) _Float16 v16h;
typedef __attribute__((ext_vector_type(8)))  float    v8f;

// Problem constants
#define BB   64
#define TT   512
#define NN   256
#define EE   16
#define TOPK 4
#define KSEAS 5

// ---------------- workspace layout (bytes) ----------------
// Fc/Fs : f16 [256][512]            -> 262144 each
// Re/Im : f32 [B][256][256]         -> 16777216 each
// sidx  : i32 [B][N][5]             -> 327680
// sre/sim: f32 [B][N][5]            -> 327680 each
// xr    : f32 [B][T]                -> 131072
// part  : f32 [16][B][T]            -> 2097152
// prob  : f32 [B][E]                -> 4096
static const size_t OFF_FC   = 0;
static const size_t OFF_FS   = OFF_FC   + 262144;
static const size_t OFF_RE   = OFF_FS   + 262144;
static const size_t OFF_IM   = OFF_RE   + 16777216;
static const size_t OFF_SIDX = OFF_IM   + 16777216;
static const size_t OFF_SRE  = OFF_SIDX + 327680;
static const size_t OFF_SIM  = OFF_SRE  + 327680;
static const size_t OFF_XR   = OFF_SIM  + 327680;
static const size_t OFF_PART = OFF_XR   + 131072;
static const size_t OFF_PROB = OFF_PART + 2097152;

// ============================================================
// Kernel 0: build DFT factor matrices in f16 (f = 0..255, t = 0..511)
// ============================================================
__global__ void build_dft(unsigned short* Fc_, unsigned short* Fs_) {
    _Float16* Fc = reinterpret_cast<_Float16*>(Fc_);
    _Float16* Fs = reinterpret_cast<_Float16*>(Fs_);
    int f = blockIdx.x;          // 256 blocks
    int t = threadIdx.x;         // 512 threads
    int m = (f * t) & (TT - 1);  // exact phase, period 512
    float ang = (float)m * (6.28318530717958647692f / (float)TT);
    float s, c;
    __sincosf(ang, &s, &c);
    Fc[f * TT + t] = (_Float16)c;
    Fs[f * TT + t] = (_Float16)s;
}

// ============================================================
// Kernel 1: batched real DFT as WMMA GEMM.
//   Re[b][f][n] =  sum_t cos(2pi f t/T) * x[b][t][n]
//   Im[b][f][n] = -sum_t sin(2pi f t/T) * x[b][t][n]
// Block = (b, ntile of 16 columns). 8 waves x 2 f-tiles cover f=0..255.
// ============================================================
__global__ void dft_wmma(const float* __restrict__ x,
                         const unsigned short* __restrict__ Fc_,
                         const unsigned short* __restrict__ Fs_,
                         float* __restrict__ Re, float* __restrict__ Im) {
    const _Float16* Fc = reinterpret_cast<const _Float16*>(Fc_);
    const _Float16* Fs = reinterpret_cast<const _Float16*>(Fs_);

    __shared__ _Float16 xt[16][520];   // [n_local][t], padded stride vs bank aliasing

    const int b  = blockIdx.x >> 4;
    const int n0 = (blockIdx.x & 15) << 4;
    const int tid  = threadIdx.x;
    const int lane = tid & 31;
    const int wave = tid >> 5;

    // stage x[b, :, n0..n0+15] -> LDS as f16 (coalesced over n, then t)
    {
        const int nl = tid & 15;
        const int tr = tid >> 4;          // 0..15
        const size_t base = ((size_t)b * TT) * NN + n0 + nl;
        for (int i = 0; i < 32; ++i) {
            int t = tr + 16 * i;
            xt[nl][t] = (_Float16)x[base + (size_t)t * NN];
        }
    }
    __syncthreads();

    const int n_l   = lane & 15;
    const int khalf = (lane >= 16) ? 8 : 0;    // A-frag K sub-offset
    const int koff  = (lane >= 16) ? 8 : 0;    // B-frag K sub-offset (same table order as A)

    for (int rep = 0; rep < 2; ++rep) {
        const int ftile = wave + 8 * rep;      // 16 f-tiles of 16 rows
        const int m = ftile * 16 + n_l;        // A-matrix row (frequency)

        v8f accR = {};
        v8f accI = {};

        for (int k0 = 0; k0 < TT; k0 += 32) {
            const _Float16* fcRow = Fc + (size_t)m * TT + k0 + khalf;
            const _Float16* fsRow = Fs + (size_t)m * TT + k0 + khalf;
            v16h ac, as, bf;
            #pragma unroll
            for (int j = 0; j < 16; ++j) {
                int kk = (j < 8) ? j : (8 + j);   // {0..7} then {16..23}
                ac[j] = fcRow[kk];
                as[j] = fsRow[kk];
            }
            #pragma unroll
            for (int j = 0; j < 16; ++j)
                bf[j] = xt[n_l][k0 + koff + ((j < 8) ? j : (8 + j))];

            accR = __builtin_amdgcn_wmma_f32_16x16x32_f16(
                       false, ac, false, bf, (short)0, accR, false, false);
            accI = __builtin_amdgcn_wmma_f32_16x16x32_f16(
                       false, as, false, bf, (short)0, accI, false, false);
        }

        const int mbase = ftile * 16 + ((lane >= 16) ? 8 : 0);
        const int n     = n0 + n_l;
        #pragma unroll
        for (int r = 0; r < 8; ++r) {
            size_t idx = ((size_t)b * 256 + (mbase + r)) * NN + n;
            Re[idx] = accR[r];
            Im[idx] = -accI[r];
        }
    }
}

// ============================================================
// Kernel 2: per (b,n) top-5 frequencies by |X|^2 over f = 1..255
// ============================================================
__global__ void topk_seas(const float* __restrict__ Re, const float* __restrict__ Im,
                          int* __restrict__ sidx, float* __restrict__ sre,
                          float* __restrict__ sim) {
    int g = blockIdx.x * blockDim.x + threadIdx.x;   // 16384 threads
    int b = g >> 8;
    int n = g & 255;

    float bm[KSEAS], br[KSEAS], bi[KSEAS];
    int   bx[KSEAS];
    #pragma unroll
    for (int k = 0; k < KSEAS; ++k) { bm[k] = -1.0f; br[k] = 0.f; bi[k] = 0.f; bx[k] = 1; }

    const size_t base = (size_t)b * 256 * NN + n;
    for (int f = 1; f < 256; ++f) {
        float re = Re[base + (size_t)f * NN];
        float im = Im[base + (size_t)f * NN];
        float m2 = re * re + im * im;
        int pos = KSEAS;
        while (pos > 0 && m2 > bm[pos - 1]) --pos;   // strict > keeps earliest on ties
        if (pos < KSEAS) {
            for (int q = KSEAS - 1; q > pos; --q) {
                bm[q] = bm[q - 1]; br[q] = br[q - 1]; bi[q] = bi[q - 1]; bx[q] = bx[q - 1];
            }
            bm[pos] = m2; br[pos] = re; bi[pos] = im; bx[pos] = f;
        }
    }
    size_t obase = ((size_t)b * NN + n) * KSEAS;
    #pragma unroll
    for (int k = 0; k < KSEAS; ++k) {
        sidx[obase + k] = bx[k];
        sre [obase + k] = br[k];
        sim [obase + k] = bi[k];
    }
}

// ============================================================
// Kernel 3: trend + seasonality + start_W reduction over n
// Block = (b, 16-n tile); writes deterministic per-tile partials.
// ============================================================
__global__ void trend_reduce(const float* __restrict__ x,
                             const int* __restrict__ sidx,
                             const float* __restrict__ sre,
                             const float* __restrict__ sim,
                             const float* __restrict__ start_W,
                             const float* __restrict__ trend_W,
                             const float* __restrict__ trend_b,
                             float* __restrict__ part) {
    __shared__ float xt[16][516];
    __shared__ float s_re[16][KSEAS], s_im[16][KSEAS];
    __shared__ int   s_ix[16][KSEAS];

    const int b  = blockIdx.x >> 4;
    const int nt = blockIdx.x & 15;
    const int n0 = nt << 4;
    const int tid = threadIdx.x;

    // stage x tile
    {
        const int nl = tid & 15;
        const int tr = tid >> 4;
        const size_t base = ((size_t)b * TT) * NN + n0 + nl;
        for (int i = 0; i < 32; ++i) {
            int t = tr + 16 * i;
            xt[nl][t] = x[base + (size_t)t * NN];
        }
    }
    if (tid < 16 * KSEAS) {
        int nl = tid / KSEAS, k = tid % KSEAS;
        size_t sbase = ((size_t)b * NN + n0 + nl) * KSEAS + k;
        s_ix[nl][k] = sidx[sbase];
        s_re[nl][k] = sre[sbase];
        s_im[nl][k] = sim[sbase];
    }
    __syncthreads();

    const float tw0 = trend_W[0], tw1 = trend_W[1], tw2 = trend_W[2];
    const float tb0 = trend_b[0], tb1 = trend_b[1], tb2 = trend_b[2];

    const int lane = tid & 31;
    const int wave = tid >> 5;
    const int nl   = lane & 15;
    const int th   = lane >> 4;              // 0/1: two t's per wave
    const float wN = start_W[n0 + nl];
    const float scale = 2.0f / (float)TT;
    const float w0 = 6.28318530717958647692f / (float)TT;

    for (int it = 0; it < 32; ++it) {
        const int t = it * 16 + wave * 2 + th;
        const float xs = xt[nl][t];

        // moving averages with edge replication (fronts 2,4,6)
        float ma4 = 0.f, ma8 = 0.f, ma12 = 0.f;
        #pragma unroll
        for (int j = 0; j < 4; ++j) {
            int u = t - 2 + j; u = (u < 0) ? 0 : (u > TT - 1 ? TT - 1 : u);
            ma4 += xt[nl][u];
        }
        #pragma unroll
        for (int j = 0; j < 8; ++j) {
            int u = t - 4 + j; u = (u < 0) ? 0 : (u > TT - 1 ? TT - 1 : u);
            ma8 += xt[nl][u];
        }
        #pragma unroll
        for (int j = 0; j < 12; ++j) {
            int u = t - 6 + j; u = (u < 0) ? 0 : (u > TT - 1 ? TT - 1 : u);
            ma12 += xt[nl][u];
        }
        ma4 *= 0.25f; ma8 *= 0.125f; ma12 *= (1.0f / 12.0f);

        float l0 = xs * tw0 + tb0, l1 = xs * tw1 + tb1, l2 = xs * tw2 + tb2;
        float mx = fmaxf(l0, fmaxf(l1, l2));
        float e0 = expf(l0 - mx), e1 = expf(l1 - mx), e2 = expf(l2 - mx);
        float trend = (ma4 * e0 + ma8 * e1 + ma12 * e2) / (e0 + e1 + e2);

        float seas = 0.f;
        #pragma unroll
        for (int k = 0; k < KSEAS; ++k) {
            int m = (s_ix[nl][k] * t) & (TT - 1);    // exact period-512 reduction
            float s, c;
            __sincosf(w0 * (float)m, &s, &c);
            seas += s_re[nl][k] * c - s_im[nl][k] * s;
        }
        seas *= scale;

        float acc = (xs + seas + trend) * wN;

        // reduce over the 16 n-lanes of this half-wave
        #pragma unroll
        for (int off = 1; off < 16; off <<= 1)
            acc += __shfl_xor(acc, off, 32);

        if (nl == 0)
            part[((size_t)nt * BB + b) * TT + t] = acc;
    }
}

// ============================================================
// Kernel 4: xr = start_b + sum over 16 tiles (deterministic)
// ============================================================
__global__ void reduce_xr(const float* __restrict__ part,
                          const float* __restrict__ start_b,
                          float* __restrict__ xr) {
    int i = blockIdx.x * blockDim.x + threadIdx.x;   // 32768
    float s = start_b[0];
    #pragma unroll
    for (int nt = 0; nt < 16; ++nt)
        s += part[(size_t)nt * (BB * TT) + i];
    xr[i] = s;
}

// ============================================================
// Kernel 5: routing per batch row
// ============================================================
__global__ void routing(const float* __restrict__ xr,
                        const float* __restrict__ gate_W, const float* __restrict__ gate_b,
                        const float* __restrict__ noise_W, const float* __restrict__ noise_b,
                        const float* __restrict__ noise, const int* __restrict__ train_flag,
                        float* __restrict__ gates_out, float* __restrict__ prob) {
    __shared__ float pg[EE][16], pn[EE][16];
    __shared__ float cleanS[EE], stdS[EE], logitS[EE];

    const int b = blockIdx.x;
    const int tid = threadIdx.x;
    const int e = tid >> 4, c = tid & 15;

    float sg = 0.f, sn = 0.f;
    const float* xrb = xr + (size_t)b * TT;
    for (int j = 0; j < 32; ++j) {
        int t = c * 32 + j;
        float xv = xrb[t];
        sg += xv * gate_W [e * TT + t];
        sn += xv * noise_W[e * TT + t];
    }
    pg[e][c] = sg; pn[e][c] = sn;
    __syncthreads();

    const int train = *train_flag;
    if (tid < EE) {
        float a = 0.f, d = 0.f;
        #pragma unroll
        for (int q = 0; q < 16; ++q) { a += pg[tid][q]; d += pn[tid][q]; }
        float clean = a + gate_b[tid];
        float raw   = d + noise_b[tid];
        float sp    = (raw > 20.f) ? raw : log1pf(expf(raw));
        float sd    = sp + 0.01f;
        cleanS[tid] = clean;
        stdS[tid]   = sd;
        logitS[tid] = train ? (clean + noise[b * EE + tid] * sd) : clean;
    }
    __syncthreads();

    if (tid == 0) {
        // top-5 of 16 (descending; ties -> lower index first)
        float v[KSEAS]; int ix[KSEAS];
        #pragma unroll
        for (int k = 0; k < KSEAS; ++k) { v[k] = -1e30f; ix[k] = 0; }
        for (int q = 0; q < EE; ++q) {
            float L = logitS[q];
            int pos = KSEAS;
            while (pos > 0 && L > v[pos - 1]) --pos;
            if (pos < KSEAS) {
                for (int r = KSEAS - 1; r > pos; --r) { v[r] = v[r - 1]; ix[r] = ix[r - 1]; }
                v[pos] = L; ix[pos] = q;
            }
        }
        // softmax over top-4
        float mx = v[0], ex[TOPK], se = 0.f;
        #pragma unroll
        for (int k = 0; k < TOPK; ++k) { ex[k] = expf(v[k] - mx); se += ex[k]; }

        float* grow = gates_out + (size_t)b * EE;
        #pragma unroll
        for (int q = 0; q < EE; ++q) grow[q] = 0.f;
        #pragma unroll
        for (int k = 0; k < TOPK; ++k) grow[ix[k]] = ex[k] / se;

        float* prow = prob + (size_t)b * EE;
        if (train) {
            float thr_in = v[TOPK], thr_out = v[TOPK - 1];
            for (int q = 0; q < EE; ++q) {
                float thr = (logitS[q] > thr_in) ? thr_in : thr_out;
                float z   = (cleanS[q] - thr) / stdS[q];
                prow[q]   = 0.5f * erfcf(-z * 0.70710678118654752f);
            }
        } else {
            for (int q = 0; q < EE; ++q) prow[q] = 0.f;
            for (int k = 0; k < TOPK; ++k) prow[ix[k]] = 1.0f;
        }
    }
}

// ============================================================
// Kernel 6: load[e] = sum_b prob[b][e] (deterministic order)
// ============================================================
__global__ void load_sum(const float* __restrict__ prob, float* __restrict__ load_out) {
    int e = threadIdx.x;   // 16 threads
    float s = 0.f;
    for (int b = 0; b < BB; ++b) s += prob[b * EE + e];
    load_out[e] = s;
}

// ============================================================
extern "C" void kernel_launch(void* const* d_in, const int* in_sizes, int n_in,
                              void* d_out, int out_size, void* d_ws, size_t ws_size,
                              hipStream_t stream) {
    (void)in_sizes; (void)n_in; (void)out_size; (void)ws_size;

    const float* x       = (const float*)d_in[0];
    const float* noise   = (const float*)d_in[1];
    const float* start_W = (const float*)d_in[2];
    const float* start_b = (const float*)d_in[3];
    const float* gate_W  = (const float*)d_in[4];
    const float* gate_b  = (const float*)d_in[5];
    const float* noise_W = (const float*)d_in[6];
    const float* noise_b = (const float*)d_in[7];
    const float* trend_W = (const float*)d_in[8];
    const float* trend_b = (const float*)d_in[9];
    const int*   train   = (const int*)d_in[10];

    char* ws = (char*)d_ws;
    unsigned short* Fc  = (unsigned short*)(ws + OFF_FC);
    unsigned short* Fs  = (unsigned short*)(ws + OFF_FS);
    float* Re   = (float*)(ws + OFF_RE);
    float* Im   = (float*)(ws + OFF_IM);
    int*   sidx = (int*)  (ws + OFF_SIDX);
    float* sre  = (float*)(ws + OFF_SRE);
    float* sim  = (float*)(ws + OFF_SIM);
    float* xr   = (float*)(ws + OFF_XR);
    float* part = (float*)(ws + OFF_PART);
    float* prob = (float*)(ws + OFF_PROB);

    float* gates_out = (float*)d_out;           // (B, E) = 1024 floats
    float* load_out  = gates_out + BB * EE;     // (E,)   = 16 floats

    build_dft   <<<256,  512, 0, stream>>>(Fc, Fs);
    dft_wmma    <<<BB * 16, 256, 0, stream>>>(x, Fc, Fs, Re, Im);
    topk_seas   <<<BB,   256, 0, stream>>>(Re, Im, sidx, sre, sim);
    trend_reduce<<<BB * 16, 256, 0, stream>>>(x, sidx, sre, sim, start_W,
                                              trend_W, trend_b, part);
    reduce_xr   <<<128,  256, 0, stream>>>(part, start_b, xr);
    routing     <<<BB,   256, 0, stream>>>(xr, gate_W, gate_b, noise_W, noise_b,
                                           noise, train, gates_out, prob);
    load_sum    <<<1,     16, 0, stream>>>(prob, load_out);
}
